// Discriminator_Model_7962869367096
// MI455X (gfx1250) — matrix operand order repaired and checked by reading, not compiled, not run on hardware
//
#include <hip/hip_runtime.h>
#include <cstdint>

// Problem constants (from reference)
#define N_NODES 524288
#define N_EDGES 4194304
#define BATCH_B 16384
#define COMB_D  432
#define EPS     1e-5f
#define SLOPE   0.2f

typedef __attribute__((ext_vector_type(16))) __bf16 v16bf;
typedef __attribute__((ext_vector_type(8)))  __bf16 v8bf;
typedef __attribute__((ext_vector_type(8)))  float  v8f;

__device__ __forceinline__ __bf16 f2bf(float f) {
  union { float f; unsigned u; } v; v.f = f;
  unsigned r = (v.u + 0x7FFFu + ((v.u >> 16) & 1u)) >> 16;  // round-nearest-even
  unsigned short s = (unsigned short)r;
  return __builtin_bit_cast(__bf16, s);
}

// ---------------- utility ----------------
__global__ void k_fill(float* __restrict__ p, float v, size_t n) {
  size_t i = (size_t)blockIdx.x * blockDim.x + threadIdx.x;
  size_t stride = (size_t)gridDim.x * blockDim.x;
  for (; i < n; i += stride) p[i] = v;
}

// dst[r*Kp + k] = k<K ? bf16(src[r*ldsrc + k]) : 0   over total = rows*Kp
__global__ void k_cvt_pad(const float* __restrict__ src, int ldsrc, int K,
                          __bf16* __restrict__ dst, int Kp, size_t total) {
  size_t i = (size_t)blockIdx.x * blockDim.x + threadIdx.x;
  size_t stride = (size_t)gridDim.x * blockDim.x;
  for (; i < total; i += stride) {
    size_t r = i / (size_t)Kp;
    int k = (int)(i - r * (size_t)Kp);
    dst[i] = (k < K) ? f2bf(src[r * (size_t)ldsrc + k]) : f2bf(0.f);
  }
}

// dst[n*Kp + k] = (n<N && k<K) ? bf16(w[k*N + n]) : 0   (transpose+pad), total = Np*Kp
__global__ void k_wt(const float* __restrict__ w, int K, int N,
                     __bf16* __restrict__ dst, int Kp, size_t total) {
  size_t i = (size_t)blockIdx.x * blockDim.x + threadIdx.x;
  size_t stride = (size_t)gridDim.x * blockDim.x;
  for (; i < total; i += stride) {
    size_t n = i / (size_t)Kp;
    int k = (int)(i - n * (size_t)Kp);
    dst[i] = (n < (size_t)N && k < K) ? f2bf(w[(size_t)k * N + n]) : f2bf(0.f);
  }
}

__device__ __forceinline__ void atomicMaxFloat(float* addr, float val) {
  if (val >= 0.f) atomicMax((int*)addr, __float_as_int(val));
  else            atomicMin((unsigned int*)addr, __float_as_uint(val));
}

// ---------------- GAT kernels ----------------
template <int CIN, int COUT>
__global__ void k_gat_transform(const float* __restrict__ xin, const float* __restrict__ W,
                                const float* __restrict__ a_s, const float* __restrict__ a_d,
                                float* __restrict__ h, float* __restrict__ es,
                                float* __restrict__ ed, int n) {
  int i = blockIdx.x * blockDim.x + threadIdx.x;
  if (i >= n) return;
  float xi[CIN];
#pragma unroll
  for (int c = 0; c < CIN; c++) xi[c] = xin[(size_t)i * CIN + c];
  float s = 0.f, d = 0.f;
#pragma unroll
  for (int o = 0; o < COUT; o++) {
    float acc = 0.f;
#pragma unroll
    for (int c = 0; c < CIN; c++) acc += xi[c] * W[c * COUT + o];
    h[(size_t)i * COUT + o] = acc;
    s += acc * a_s[o];
    d += acc * a_d[o];
  }
  es[i] = s;
  ed[i] = d;
}

__global__ void k_edge_max(const int* __restrict__ ei, int E, int n,
                           const float* __restrict__ es, const float* __restrict__ ed,
                           float* __restrict__ m) {
  int t = blockIdx.x * blockDim.x + threadIdx.x;
  if (t >= E + n) return;
  int s, d;
  if (t < E) { s = ei[t]; d = ei[E + t]; } else { s = d = t - E; }
  float e = es[s] + ed[d];
  e = e > 0.f ? e : SLOPE * e;
  atomicMaxFloat(&m[d], e);
}

template <int COUT>
__global__ void k_edge_sum(const int* __restrict__ ei, int E, int n,
                           const float* __restrict__ es, const float* __restrict__ ed,
                           const float* __restrict__ m, float* __restrict__ den,
                           const float* __restrict__ h, float* __restrict__ acc) {
  int t = blockIdx.x * blockDim.x + threadIdx.x;
  if (t >= E + n) return;
  int s, d;
  if (t < E) { s = ei[t]; d = ei[E + t]; } else { s = d = t - E; }
  float e = es[s] + ed[d];
  e = e > 0.f ? e : SLOPE * e;
  float w = __expf(e - m[d]);
  atomicAdd(&den[d], w);
  const float* hs = h + (size_t)s * COUT;
  float* ad = acc + (size_t)d * COUT;
#pragma unroll
  for (int c = 0; c < COUT; c++) atomicAdd(&ad[c], w * hs[c]);
}

template <int COUT>
__global__ void k_gat_finish(const float* __restrict__ acc, const float* __restrict__ den,
                             const float* __restrict__ b, float* __restrict__ g, int n) {
  int i = blockIdx.x * blockDim.x + threadIdx.x;
  if (i >= n) return;
  float inv = 1.f / den[i];
#pragma unroll
  for (int c = 0; c < COUT; c++) {
    float v = acc[(size_t)i * COUT + c] * inv + b[c];
    g[(size_t)i * COUT + c] = v > 0.f ? v : 0.f;
  }
}

__global__ void k_pool(const float* __restrict__ acc, const float* __restrict__ den,
                       const float* __restrict__ b2, const int* __restrict__ batch,
                       float* __restrict__ comb, int n) {
  int i = blockIdx.x * blockDim.x + threadIdx.x;
  if (i >= n) return;
  float inv = 1.f / den[i];
  float* o = comb + (size_t)batch[i] * COMB_D;
#pragma unroll
  for (int c = 0; c < 32; c++) {
    float v = acc[(size_t)i * 32 + c] * inv + b2[c];
    v = v > 0.f ? v : 0.f;
    atomicAdd(&o[c], v);
  }
}

// ---------------- WMMA GEMM + BN + ReLU (bf16 staged, padded, branch-free) ----------------
// A: bf16 [M x Kp], Wt: bf16 [Np x Kp] (Wt[n*Kp+k] = W[k,n]); Kp % 64 == 0, Np % 16 == 0.
// One 16x16 C tile per wave; 4 waves per block (64 rows). M % 64 == 0.
__global__ void k_gemm_bf16(const __bf16* __restrict__ A, int Kp,
                            const __bf16* __restrict__ Wt, int Nout,
                            const float* __restrict__ bias, const float* __restrict__ bn,
                            float* __restrict__ Cout, int ldc) {
  int lane = threadIdx.x & 31;
  int wave = threadIdx.x >> 5;
  int m0 = (blockIdx.y * 4 + wave) * 16;
  int n0 = blockIdx.x * 16;
  int half = lane >> 4;                 // 0: lanes 0-15, 1: lanes 16-31

  const __bf16* arow = A + (size_t)(m0 + (lane & 15)) * Kp + half * 8;
  const __bf16* bcol = Wt + (size_t)(n0 + (lane & 15)) * Kp + half * 8;

  v8f acc = {};
  for (int k0 = 0; k0 < Kp; k0 += 64) {
    __builtin_prefetch(arow + k0 + 128, 0, 3);   // -> global_prefetch_b8
    v8bf a0 = *(const v8bf*)(arow + k0);
    v8bf a1 = *(const v8bf*)(arow + k0 + 16);
    v8bf b0 = *(const v8bf*)(bcol + k0);
    v8bf b1 = *(const v8bf*)(bcol + k0 + 16);
    v16bf av = __builtin_shufflevector(a0, a1, 0,1,2,3,4,5,6,7,8,9,10,11,12,13,14,15);
    v16bf bv = __builtin_shufflevector(b0, b1, 0,1,2,3,4,5,6,7,8,9,10,11,12,13,14,15);
    acc = __builtin_amdgcn_wmma_f32_16x16x32_bf16(false, av, false, bv,
                                                  (short)0, acc, false, false);
    a0 = *(const v8bf*)(arow + k0 + 32);
    a1 = *(const v8bf*)(arow + k0 + 48);
    b0 = *(const v8bf*)(bcol + k0 + 32);
    b1 = *(const v8bf*)(bcol + k0 + 48);
    av = __builtin_shufflevector(a0, a1, 0,1,2,3,4,5,6,7,8,9,10,11,12,13,14,15);
    bv = __builtin_shufflevector(b0, b1, 0,1,2,3,4,5,6,7,8,9,10,11,12,13,14,15);
    acc = __builtin_amdgcn_wmma_f32_16x16x32_bf16(false, av, false, bv,
                                                  (short)0, acc, false, false);
  }

  int orow = m0 + half * 8;             // C/D layout: vgpr r -> M = half*8 + r
  int ocol = n0 + (lane & 15);
  if (ocol < Nout) {
    float bi = bias[ocol];
    float gamma = bn[ocol], beta = bn[Nout + ocol];
    float mean = bn[2 * Nout + ocol], var = bn[3 * Nout + ocol];
    float scale = gamma * rsqrtf(var + EPS);
    float shift = beta - mean * scale;
#pragma unroll
    for (int r = 0; r < 8; r++) {
      float v = (acc[r] + bi) * scale + shift;
      Cout[(size_t)(orow + r) * ldc + ocol] = v > 0.f ? v : 0.f;
    }
  }
}

__global__ void k_final(const float* __restrict__ hc, const float* __restrict__ w,
                        const float* __restrict__ b, float* __restrict__ out, int B) {
  int i = blockIdx.x * blockDim.x + threadIdx.x;
  if (i >= B) return;
  float s = b[0];
#pragma unroll 4
  for (int c = 0; c < 100; c++) s += hc[(size_t)i * 100 + c] * w[c];
  out[i] = s;
}

// ---------------- host ----------------
extern "C" void kernel_launch(void* const* d_in, const int* in_sizes, int n_in,
                              void* d_out, int out_size, void* d_ws, size_t ws_size,
                              hipStream_t stream) {
  const float* x     = (const float*)d_in[0];
  const int*   ei    = (const int*)d_in[1];
  const int*   batch = (const int*)d_in[2];
  const float* fing  = (const float*)d_in[3];
  const float* tvin  = (const float*)d_in[4];
  const float* W1    = (const float*)d_in[5];
  const float* as1   = (const float*)d_in[6];
  const float* ad1   = (const float*)d_in[7];
  const float* b1    = (const float*)d_in[8];
  const float* W2    = (const float*)d_in[9];
  const float* as2   = (const float*)d_in[10];
  const float* ad2   = (const float*)d_in[11];
  const float* b2    = (const float*)d_in[12];
  const float* fc1w  = (const float*)d_in[13];
  const float* fc1b  = (const float*)d_in[14];
  const float* bn1   = (const float*)d_in[15];
  const float* fc2w  = (const float*)d_in[16];
  const float* fc2b  = (const float*)d_in[17];
  const float* bn2   = (const float*)d_in[18];
  const float* fc3w  = (const float*)d_in[19];
  const float* fc3b  = (const float*)d_in[20];
  const float* bn3   = (const float*)d_in[21];
  const float* fc4w  = (const float*)d_in[22];
  const float* fc4b  = (const float*)d_in[23];
  const float* bn4   = (const float*)d_in[24];
  const float* fcc1w = (const float*)d_in[25];
  const float* fcc1b = (const float*)d_in[26];
  const float* bnc   = (const float*)d_in[27];
  const float* fcc2w = (const float*)d_in[28];
  const float* fcc2b = (const float*)d_in[29];
  float* out = (float*)d_out;

  // Padded GEMM geometry (Kp % 64, Np % 16)
  const int Kp1 = 1024, Np1 = 400;   // fc1: 1024 -> 400
  const int Kp2 = 448,  Np2 = 208;   // fc2: 400  -> 200
  const int Kp3 = 256,  Np3 = 400;   // fc3: 256  -> 400
  const int Kp4 = 448,  Np4 = 208;   // fc4: 400  -> 200
  const int Kpc = 448,  Npc = 112;   // fcc1: 432 -> 100

  // workspace layout (floats)
  float* ws = (float*)d_ws;
  size_t o = 0;
  float* comb = ws + o; o += (size_t)BATCH_B * COMB_D;     // [B,432] pool|fp|tv (fp32)
  float* hc   = ws + o; o += (size_t)BATCH_B * 100;        // fcc1 out (fp32)
  __bf16* abf = (__bf16*)(ws + o); o += (size_t)BATCH_B * 1024 / 2;  // bf16 A staging (max Kp)
  __bf16* wt1 = (__bf16*)(ws + o);
  __bf16* wt2 = wt1 + (size_t)Np1 * Kp1;
  __bf16* wt3 = wt2 + (size_t)Np2 * Kp2;
  __bf16* wt4 = wt3 + (size_t)Np3 * Kp3;
  __bf16* wtc = wt4 + (size_t)Np4 * Kp4;
  size_t wtot = (size_t)Np1*Kp1 + (size_t)Np2*Kp2 + (size_t)Np3*Kp3 + (size_t)Np4*Kp4 + (size_t)Npc*Kpc;
  o += (wtot + 3) / 2;                                     // bf16 -> float units (rounded)
  float* base = ws + o;
  float* h1   = base;                                      // [N,14]
  float* acc1 = h1 + (size_t)N_NODES * 14;                 // [N,14]
  float* g1   = acc1 + (size_t)N_NODES * 14;               // [N,14]
  float* es   = g1 + (size_t)N_NODES * 14;                 // [N]
  float* ed   = es + N_NODES;
  float* mm   = ed + N_NODES;
  float* den  = mm + N_NODES;
  float* h2   = den + N_NODES;                             // [N,32]
  float* acc2 = h1;       // aliases h1+acc1+g1 (N*42 >= N*32), all dead by then
  float* tmp  = base;     // fp32 MLP temp [B,400]; runs before GNN region is live

  size_t need = (o + (size_t)N_NODES * (14 * 3 + 4 + 32)) * sizeof(float);
  if (ws_size < need) return;

  const int TB = 256;
  const int Etot = N_EDGES + N_NODES;
  dim3 nb((N_NODES + TB - 1) / TB), ebk((Etot + TB - 1) / TB), tb(TB);
  auto blks = [&](size_t n) {
    size_t b = (n + TB - 1) / TB;
    return (int)(b > 16384 ? 16384 : b);
  };
  auto fill = [&](float* p, float v, size_t n) {
    k_fill<<<blks(n), TB, 0, stream>>>(p, v, n);
  };
  auto cvt = [&](const float* src, int ldsrc, int K, __bf16* dst, int Kp) {
    size_t total = (size_t)BATCH_B * Kp;
    k_cvt_pad<<<blks(total), TB, 0, stream>>>(src, ldsrc, K, dst, Kp, total);
  };
  auto wt = [&](const float* w, int K, int N, __bf16* dst, int Kp, int Np) {
    size_t total = (size_t)Np * Kp;
    k_wt<<<blks(total), TB, 0, stream>>>(w, K, N, dst, Kp, total);
  };
  auto gemm = [&](const __bf16* A, int Kp, const __bf16* Wt, int Np, int Nout,
                  const float* bias, const float* bn, float* C, int ldc) {
    dim3 g(Np / 16, BATCH_B / 64);
    k_gemm_bf16<<<g, dim3(128), 0, stream>>>(A, Kp, Wt, Nout, bias, bn, C, ldc);
  };

  // ---- weight staging (transpose + pad to bf16) ----
  wt(fc1w, 1024, 400, wt1, Kp1, Np1);
  wt(fc2w, 400, 200, wt2, Kp2, Np2);
  wt(fc3w, 256, 400, wt3, Kp3, Np3);
  wt(fc4w, 400, 200, wt4, Kp4, Np4);
  wt(fcc1w, 432, 100, wtc, Kpc, Npc);

  fill(comb, 0.f, (size_t)BATCH_B * COMB_D);

  // ---- MLP branches (WMMA), before GNN region is touched ----
  cvt(fing, 1024, 1024, abf, Kp1);
  gemm(abf, Kp1, wt1, Np1, 400, fc1b, bn1, tmp, 400);            // fp1
  cvt(tmp, 400, 400, abf, Kp2);
  gemm(abf, Kp2, wt2, Np2, 200, fc2b, bn2, comb + 32, COMB_D);   // fp2 -> comb[:,32:232]
  cvt(tvin, 256, 256, abf, Kp3);
  gemm(abf, Kp3, wt3, Np3, 400, fc3b, bn3, tmp, 400);            // tv1
  cvt(tmp, 400, 400, abf, Kp4);
  gemm(abf, Kp4, wt4, Np4, 200, fc4b, bn4, comb + 232, COMB_D);  // tv2 -> comb[:,232:432]

  // ---- GAT layer 1 (14 -> 14) ----
  k_gat_transform<14, 14><<<nb, tb, 0, stream>>>(x, W1, as1, ad1, h1, es, ed, N_NODES);
  fill(mm, -INFINITY, N_NODES);
  fill(den, 0.f, N_NODES);
  fill(acc1, 0.f, (size_t)N_NODES * 14);
  k_edge_max<<<ebk, tb, 0, stream>>>(ei, N_EDGES, N_NODES, es, ed, mm);
  k_edge_sum<14><<<ebk, tb, 0, stream>>>(ei, N_EDGES, N_NODES, es, ed, mm, den, h1, acc1);
  k_gat_finish<14><<<nb, tb, 0, stream>>>(acc1, den, b1, g1, N_NODES);

  // ---- GAT layer 2 (14 -> 32) ----
  k_gat_transform<14, 32><<<nb, tb, 0, stream>>>(g1, W2, as2, ad2, h2, es, ed, N_NODES);
  fill(mm, -INFINITY, N_NODES);
  fill(den, 0.f, N_NODES);
  fill(acc2, 0.f, (size_t)N_NODES * 32);
  k_edge_max<<<ebk, tb, 0, stream>>>(ei, N_EDGES, N_NODES, es, ed, mm);
  k_edge_sum<32><<<ebk, tb, 0, stream>>>(ei, N_EDGES, N_NODES, es, ed, mm, den, h2, acc2);
  k_pool<<<nb, tb, 0, stream>>>(acc2, den, b2, batch, comb, N_NODES);

  // ---- combined head ----
  cvt(comb, COMB_D, COMB_D, abf, Kpc);
  gemm(abf, Kpc, wtc, Npc, 100, fcc1b, bnc, hc, 100);            // WMMA
  k_final<<<(BATCH_B + TB - 1) / TB, tb, 0, stream>>>(hc, fcc2w, fcc2b, out, BATCH_B);
}
